// ALiBi_76278619176977
// MI455X (gfx1250) — matrix operand order repaired and checked by reading, not compiled, not run on hardware
//
#include <hip/hip_runtime.h>

// CDNA5 / gfx1250 two-phase flash attention with distance-bias epilogue.
//   convert_k / convert_v : f32 -> f16 into WMMA-fragment-ordered workspace
//   stats_kernel (pass A) : S^T = K*Q^T (v_wmma_f32_16x16x32_f16), online softmax
//                           stats; stores c = m + log2(L) per query (exp2 domain)
//   attn_out_kernel (pass B): recompute S^T, combined weight
//                           w = attn ? 0 : (exp2(s-c) - (alibi ? 0 : dist)),
//                           single PV accumulator, direct store.
// All exponentials in exp2 domain: log2(e) folded into the Q scale.

typedef _Float16 v16h __attribute__((ext_vector_type(16)));
typedef float    v8f  __attribute__((ext_vector_type(8)));
typedef float    v16f __attribute__((ext_vector_type(16)));

#define NB    16
#define NQ    2048
#define NKEY  2048
#define NF    128
#define NKT   64   // NKEY / 32 keys per iteration
// (1/sqrt(128)) * log2(e): logits produced directly in exp2 domain.
#define QK_SCALE2 (0.08838834764831843f * 1.44269504088896340f)

__device__ __forceinline__ float bperm_f(int src_lane, float v) {
  return __builtin_bit_cast(
      float, __builtin_amdgcn_ds_bpermute(src_lane << 2, __builtin_bit_cast(int, v)));
}

__device__ __forceinline__ float fast_exp2(float x) {
#if __has_builtin(__builtin_amdgcn_exp2f)
  return __builtin_amdgcn_exp2f(x);
#else
  return exp2f(x);
#endif
}

__device__ __forceinline__ v8f wmma_f16(v16h a, v16h b, v8f c) {
  // (neg_a, A, neg_b, B, c_mod, C, reuse_a, reuse_b)
  return __builtin_amdgcn_wmma_f32_16x16x32_f16(false, a, false, b, (short)0, c, false, false);
}

// Load Q rows as B-fragments (lane = query column, 16 contiguous f per lane).
__device__ __forceinline__ void load_q_frags(const float* __restrict__ q, int qrow_i,
                                             int hseg16, v16h aq[4]) {
  const float* qrow = q + (size_t)qrow_i * NF;
#pragma unroll
  for (int fr = 0; fr < 4; ++fr) {
    v16f qv = *(const v16f*)(qrow + fr * 32 + hseg16);
#pragma unroll
    for (int j = 0; j < 16; ++j) aq[fr][j] = (_Float16)(qv[j] * QK_SCALE2);
  }
}

// ---------------------------------------------------------------------------
// Pre-pass 1: K (f32 [b][key][f]) -> f16 A-fragments.
// Fragment index = (b*64 + kt)*8 + t*4 + fr ; lane l: key = kt*32+t*16+(l&15),
// f = fr*32 + hb + j, hb = (l&16)?16:0.
// ---------------------------------------------------------------------------
__global__ __launch_bounds__(256) void convert_k_kernel(
    const float* __restrict__ kin, _Float16* __restrict__ kws) {
  const int tid  = blockIdx.x * 256 + threadIdx.x;
  const int lane = tid & 31;
  const int fr   = (tid >> 5) & 3;
  const int t    = (tid >> 7) & 1;
  const int kt   = (tid >> 8) & 63;
  const int b    =  tid >> 14;
  const int key  = kt * 32 + t * 16 + (lane & 15);
  const int hb   = (lane & 16) ? 16 : 0;
  const float* src = kin + ((size_t)(b * NKEY + key)) * NF + fr * 32 + hb;
  v8f lo = *(const v8f*)(src);
  v8f hi = *(const v8f*)(src + 8);
  v16h val;
#pragma unroll
  for (int j = 0; j < 8; ++j) {
    val[j]     = (_Float16)lo[j];
    val[j + 8] = (_Float16)hi[j];
  }
  *(v16h*)(kws + (size_t)tid * 16) = val;
}

// ---------------------------------------------------------------------------
// Pre-pass 2: V (f32 [b][key][f]) -> f16 B-fragments.
// Fragment index = (b*64 + kt)*8 + ft ; lane l: f = ft*16 + (l&15),
// key = kt*32 + ((l&16)?8:0) + (j<8 ? j : j+8).
// ---------------------------------------------------------------------------
__global__ __launch_bounds__(256) void convert_v_kernel(
    const float* __restrict__ vin, _Float16* __restrict__ vws) {
  const int tid  = blockIdx.x * 256 + threadIdx.x;
  const int lane = tid & 31;
  const int ft   = (tid >> 5) & 7;
  const int kt   = (tid >> 8) & 63;
  const int b    =  tid >> 14;
  const int f    = ft * 16 + (lane & 15);
  const int keyb = kt * 32 + ((lane & 16) ? 8 : 0);
  const float* src = vin + ((size_t)(b * NKEY + keyb)) * NF + f;
  v16h val;
#pragma unroll
  for (int j = 0; j < 16; ++j) val[j] = (_Float16)src[(size_t)(j < 8 ? j : j + 8) * NF];
  *(v16h*)(vws + (size_t)tid * 16) = val;
}

// ---------------------------------------------------------------------------
// Pass A: softmax stats. Block = 8 waves, wave = 16 queries. Grid (NQ/128, NB).
// Stores c = m + log2(L) (exp2 domain) per query.
// ---------------------------------------------------------------------------
__global__ __launch_bounds__(256) void stats_kernel(
    const float* __restrict__ q, const v16h* __restrict__ kws,
    float* __restrict__ c_ws) {
  const int lane   = threadIdx.x & 31;
  const int wave   = threadIdx.x >> 5;
  const int b      = blockIdx.y;
  const int qbase  = blockIdx.x * 128 + wave * 16;
  const int lhalf  = lane & 15;
  const int hseg16 = (lane & 16) ? 16 : 0;
  const int qrow_i = b * NQ + qbase + lhalf;

  v16h aq[4];
  load_q_frags(q, qrow_i, hseg16, aq);

  float m_run = -__builtin_inff();
  float l_run = 0.0f;
  const size_t fgbase = (size_t)(b * NKT) * 8;

  for (int kt = 0; kt < NKT; ++kt) {
    v8f st0 = {}, st1 = {};
    const v16h* kp = kws + (fgbase + (size_t)kt * 8) * 32 + lane;
#pragma unroll
    for (int fr = 0; fr < 4; ++fr) st0 = wmma_f16(kp[(size_t)fr * 32], aq[fr], st0);
#pragma unroll
    for (int fr = 0; fr < 4; ++fr) st1 = wmma_f16(kp[(size_t)(4 + fr) * 32], aq[fr], st1);

    float tmax = st0[0];
#pragma unroll
    for (int r = 1; r < 8; ++r) tmax = fmaxf(tmax, st0[r]);
#pragma unroll
    for (int r = 0; r < 8; ++r) tmax = fmaxf(tmax, st1[r]);
    tmax = fmaxf(tmax, bperm_f(lane ^ 16, tmax));
    const float m_new = fmaxf(m_run, tmax);
    const float sf = fast_exp2(m_run - m_new);

    float tsum = 0.0f;
#pragma unroll
    for (int r = 0; r < 8; ++r) tsum += fast_exp2(st0[r] - m_new) + fast_exp2(st1[r] - m_new);
    tsum += bperm_f(lane ^ 16, tsum);
    l_run = l_run * sf + tsum;
    m_run = m_new;
  }

  if (lane < 16) c_ws[qrow_i] = m_run + log2f(l_run);
}

// ---------------------------------------------------------------------------
// Pass B: output. Combined weight w = attn ? 0 : (exp2(s - c) - (alibi ? 0 : d)).
// Single f32 accumulator, already normalized -> direct store.
// ---------------------------------------------------------------------------
__global__ __launch_bounds__(256) void attn_out_kernel(
    const float* __restrict__ q,
    const float* __restrict__ coords_q,
    const float* __restrict__ coords_k,
    const unsigned char* __restrict__ attn_mask,   // jnp.bool_ -> 1 byte
    const unsigned char* __restrict__ alibi_mask,  // jnp.bool_ -> 1 byte
    const float* __restrict__ bias_scale,
    const float* __restrict__ running_mean,
    const v16h* __restrict__ kws,
    const v16h* __restrict__ vws,
    const float* __restrict__ c_ws,
    float* __restrict__ out) {
  const int lane   = threadIdx.x & 31;
  const int wave   = threadIdx.x >> 5;
  const int b      = blockIdx.y;
  const int qbase  = blockIdx.x * 128 + wave * 16;
  const int lhalf  = lane & 15;
  const int hb     = (lane & 16) ? 8 : 0;
  const int hseg16 = (lane & 16) ? 16 : 0;
  const int qrow_i = b * NQ + qbase + lhalf;

  const float distmul = bias_scale[0] / running_mean[0];
  const float cqx = coords_q[2 * qrow_i + 0];
  const float cqy = coords_q[2 * qrow_i + 1];
  const float cstat = c_ws[qrow_i];  // m + log2(L)

  v16h aq[4];
  load_q_frags(q, qrow_i, hseg16, aq);

  v8f acc[8];
#pragma unroll
  for (int ft = 0; ft < 8; ++ft) acc[ft] = {};

  const size_t amrow  = (size_t)qrow_i * NKEY;
  const size_t fgbase = (size_t)(b * NKT) * 8;

  for (int kt = 0; kt < NKT; ++kt) {
    const int kb = kt * 32;

    // Prefetch the streaming mask rows ahead (global_prefetch_b8).
    __builtin_prefetch(attn_mask + amrow + kb + 512, 0, 1);
    __builtin_prefetch(alibi_mask + amrow + kb + 512, 0, 1);

    // ---- S^T = K_tile(32x128) * Q^T : two 16x16 f32 tiles
    v8f st0 = {}, st1 = {};
    const v16h* kp = kws + (fgbase + (size_t)kt * 8) * 32 + lane;
#pragma unroll
    for (int fr = 0; fr < 4; ++fr) st0 = wmma_f16(kp[(size_t)fr * 32], aq[fr], st0);
#pragma unroll
    for (int fr = 0; fr < 4; ++fr) st1 = wmma_f16(kp[(size_t)(4 + fr) * 32], aq[fr], st1);

    // ---- combined weight fragment (pure per-lane mapping from S^T layout)
    const unsigned long long am0 = *(const unsigned long long*)(attn_mask + amrow + kb + hb);
    const unsigned long long am1 = *(const unsigned long long*)(attn_mask + amrow + kb + 16 + hb);
    const unsigned long long al0 = *(const unsigned long long*)(alibi_mask + amrow + kb + hb);
    const unsigned long long al1 = *(const unsigned long long*)(alibi_mask + amrow + kb + 16 + hb);
    const float2* ck0 = (const float2*)(coords_k + 2 * ((size_t)b * NKEY + kb + hb));
    const float2* ck1 = (const float2*)(coords_k + 2 * ((size_t)b * NKEY + kb + 16 + hb));

    v16h w;
#pragma unroll
    for (int j = 0; j < 8; ++j) {
      const float p0 = fast_exp2(st0[j] - cstat);  // exp(s-m)/L, already normalized
      const float p1 = fast_exp2(st1[j] - cstat);
      const float2 c0 = ck0[j];
      const float2 c1 = ck1[j];
      const float dx0 = cqx - c0.x, dy0 = cqy - c0.y;
      const float dx1 = cqx - c1.x, dy1 = cqy - c1.y;
      const float d0 = sqrtf(fmaf(dx0, dx0, dy0 * dy0)) * distmul;
      const float d1 = sqrtf(fmaf(dx1, dx1, dy1 * dy1)) * distmul;
      const bool att0 = (am0 >> (8 * j)) & 1;
      const bool att1 = (am1 >> (8 * j)) & 1;
      const bool ali0 = (al0 >> (8 * j)) & 1;
      const bool ali1 = (al1 >> (8 * j)) & 1;
      const float w0 = att0 ? 0.0f : (p0 - (ali0 ? 0.0f : d0));
      const float w1 = att1 ? 0.0f : (p1 - (ali1 ? 0.0f : d1));
      w[j]     = (_Float16)w0;
      w[j + 8] = (_Float16)w1;
    }

    // ---- PV accumulation: out_tile += W(16x32) * V(32x16) per f-tile
    const v16h* vp = vws + (fgbase + (size_t)kt * 8) * 32 + lane;
#pragma unroll
    for (int ft = 0; ft < 8; ++ft) acc[ft] = wmma_f16(w, vp[(size_t)ft * 32], acc[ft]);
  }

  // ---- epilogue: already normalized, direct store
#pragma unroll
  for (int r = 0; r < 8; ++r) {
    float* op = out + ((size_t)(b * NQ) + qbase + r + hb) * NF + lhalf;
#pragma unroll
    for (int ft = 0; ft < 8; ++ft) op[ft * 16] = acc[ft][r];
  }
}

// ---------------------------------------------------------------------------
extern "C" void kernel_launch(void* const* d_in, const int* in_sizes, int n_in,
                              void* d_out, int out_size, void* d_ws, size_t ws_size,
                              hipStream_t stream) {
  (void)in_sizes; (void)n_in; (void)out_size; (void)ws_size;
  const float* q            = (const float*)d_in[0];
  const float* k            = (const float*)d_in[1];
  const float* v            = (const float*)d_in[2];
  const float* coords_q     = (const float*)d_in[3];
  const float* coords_k     = (const float*)d_in[4];
  const unsigned char* am   = (const unsigned char*)d_in[5];  // bool masks: 1 byte/elem
  const unsigned char* al   = (const unsigned char*)d_in[6];
  const float* bias_scale   = (const float*)d_in[7];
  const float* running_mean = (const float*)d_in[8];
  float* out = (float*)d_out;

  // Workspace: 8 MB f16 K frags + 8 MB f16 V frags + 128 KB softmax stats.
  _Float16* kws = (_Float16*)d_ws;
  _Float16* vws = kws + (size_t)NB * NKEY * NF;
  float*    cws = (float*)(vws + (size_t)NB * NKEY * NF);

  convert_k_kernel<<<1024, 256, 0, stream>>>(k, kws);
  convert_v_kernel<<<1024, 256, 0, stream>>>(v, vws);

  stats_kernel<<<dim3(NQ / 128, NB), 256, 0, stream>>>(q, (const v16h*)kws, cws);

  attn_out_kernel<<<dim3(NQ / 128, NB), 256, 0, stream>>>(
      q, coords_q, coords_k, am, al, bias_scale, running_mean,
      (const v16h*)kws, (const v16h*)vws, cws, out);
}
